// DotAttention_56453050139075
// MI455X (gfx1250) — compile-verified, matrix operand order checked by reading, not run on hardware
//
#include <hip/hip_runtime.h>

// Problem sizes (fixed by the reference).
#define NQ_   8192
#define NR_   8192
#define DQ_   2048
#define DR_   2048
#define DOUT_ 2048

typedef __attribute__((ext_vector_type(16))) _Float16 v16h;
typedef __attribute__((ext_vector_type(8)))  _Float16 v8h;
typedef __attribute__((ext_vector_type(8)))  float    v8f;
typedef __attribute__((ext_vector_type(4)))  float    v4f;

// ---------------------------------------------------------------------------
// Fragment loaders for V_WMMA_F32_16X16X32_F16.
//
// 16-bit A-matrix 16x32 layout (ISA 7.12.2): lane L holds row (L&15);
//   elements 0..7  = K = k0 + 8*(L>>4) + {0..7}
//   elements 8..15 = K = k0 + 16 + 8*(L>>4) + {0..7}
// B (32x16, K-major per column) uses the symmetric layout with lane = column.
// Caller passes p = rowptr + k0 + 8*(lane>>4); we read p[0..7] and p[16..23].
// ---------------------------------------------------------------------------
__device__ __forceinline__ v16h frag_load(const _Float16* __restrict__ p) {
  v8h lo = *(const v8h*)(p);
  v8h hi = *(const v8h*)(p + 16);
  return __builtin_shufflevector(lo, hi, 0,1,2,3,4,5,6,7,8,9,10,11,12,13,14,15);
}

__device__ __forceinline__ v16h frag_load(const float* __restrict__ p) {
  v4f a = *(const v4f*)(p);
  v4f b = *(const v4f*)(p + 4);
  v4f c = *(const v4f*)(p + 16);
  v4f d = *(const v4f*)(p + 20);
  v8f lo = __builtin_shufflevector(a, b, 0,1,2,3,4,5,6,7);
  v8f hi = __builtin_shufflevector(c, d, 0,1,2,3,4,5,6,7);
  v8h l8 = __builtin_convertvector(lo, v8h);
  v8h h8 = __builtin_convertvector(hi, v8h);
  return __builtin_shufflevector(l8, h8, 0,1,2,3,4,5,6,7,8,9,10,11,12,13,14,15);
}

// ---------------------------------------------------------------------------
// NT GEMM: C[M,N] = op( A[M,K] (row-major) * B[N,K]^T (row-major) )
// Block = 128 threads = 4 waves; block tile 128x128; each wave a 64x64
// quadrant = 4x4 grid of 16x16 WMMA tiles, K stepped by 32.
// ---------------------------------------------------------------------------
template<typename TIn, bool RELU, bool OUT_F16>
__global__ __launch_bounds__(128)
void gemm_nt_wmma(const TIn* __restrict__ A, const TIn* __restrict__ B,
                  void* __restrict__ Cout, int M, int N, int K, float scale) {
  const int lane = threadIdx.x & 31;
  const int wave = threadIdx.x >> 5;
  const int r16  = lane & 15;
  const int sel  = lane >> 4;                 // which K-half this lane holds

  const int row0 = blockIdx.x * 128 + (wave >> 1) * 64;   // M base of wave tile
  const int col0 = blockIdx.y * 128 + (wave & 1) * 64;    // N base of wave tile

  v8f acc[4][4] = {};

  const size_t ldk = (size_t)K;
  const TIn* arow[4];
  const TIn* brow[4];
  #pragma unroll
  for (int t = 0; t < 4; ++t) {
    arow[t] = A + (size_t)(row0 + 16 * t + r16) * ldk + sel * 8;
    brow[t] = B + (size_t)(col0 + 16 * t + r16) * ldk + sel * 8;
  }

  for (int k0 = 0; k0 < K; k0 += 32) {
    v16h af[4], bf[4];
    #pragma unroll
    for (int t = 0; t < 4; ++t) af[t] = frag_load(arow[t] + k0);
    #pragma unroll
    for (int t = 0; t < 4; ++t) bf[t] = frag_load(brow[t] + k0);

    #pragma unroll
    for (int i = 0; i < 4; ++i)
      #pragma unroll
      for (int j = 0; j < 4; ++j)
        acc[i][j] = __builtin_amdgcn_wmma_f32_16x16x32_f16(
            /*neg_a=*/false, af[i], /*neg_b=*/false, bf[j],
            /*c_mod=*/(short)0, acc[i][j], /*reuse_a=*/false, /*reuse_b=*/false);
  }

  // Epilogue. D layout: vgpr v of lane L -> row = 16*i + v + 8*(L>>4), col = 16*j + (L&15).
  if constexpr (OUT_F16) {
    _Float16* C = (_Float16*)Cout;
    #pragma unroll
    for (int i = 0; i < 4; ++i) {
      #pragma unroll
      for (int v = 0; v < 8; ++v) {
        const size_t r = (size_t)(row0 + 16 * i + 8 * sel + v);
        #pragma unroll
        for (int j = 0; j < 4; ++j) {
          float x = acc[i][j][v] * scale;
          if (RELU) x = fmaxf(x, 0.0f);
          C[r * (size_t)N + (size_t)(col0 + 16 * j + r16)] = (_Float16)x;
        }
      }
    }
  } else {
    float* C = (float*)Cout;
    #pragma unroll
    for (int i = 0; i < 4; ++i) {
      #pragma unroll
      for (int v = 0; v < 8; ++v) {
        const size_t r = (size_t)(row0 + 16 * i + 8 * sel + v);
        #pragma unroll
        for (int j = 0; j < 4; ++j) {
          float x = acc[i][j][v] * scale;
          if (RELU) x = fmaxf(x, 0.0f);
          C[r * (size_t)N + (size_t)(col0 + 16 * j + r16)] = x;
        }
      }
    }
  }
}

// ---------------------------------------------------------------------------
// Tiled transpose with fp32 -> f16 downconvert: out[c, j] = (f16) in[j, c].
// in: rows x cols (row-major f32); out: cols x rows (row-major f16).
// ---------------------------------------------------------------------------
__global__ __launch_bounds__(256)
void transpose_cvt(const float* __restrict__ in, _Float16* __restrict__ out,
                   int rows, int cols) {
  __shared__ float tile[32][33];
  const int c0 = blockIdx.x * 32;
  const int r0 = blockIdx.y * 32;
  const int tx = threadIdx.x & 31;
  const int ty = threadIdx.x >> 5;   // 8 sub-rows

  #pragma unroll
  for (int i = ty; i < 32; i += 8)
    tile[i][tx] = in[(size_t)(r0 + i) * cols + (c0 + tx)];
  __syncthreads();
  #pragma unroll
  for (int i = ty; i < 32; i += 8)
    out[(size_t)(c0 + i) * rows + (r0 + tx)] = (_Float16)tile[tx][i];
}

// ---------------------------------------------------------------------------
// In-place row softmax over f16 logits (f32 math). One block per row.
// ---------------------------------------------------------------------------
__global__ __launch_bounds__(256)
void softmax_rows(_Float16* __restrict__ S, int ncols) {
  __shared__ float red[256];
  _Float16* p = S + (size_t)blockIdx.x * (size_t)ncols;
  const int tid = threadIdx.x;

  float m = -3.0e38f;
  for (int j = tid; j < ncols; j += 256) m = fmaxf(m, (float)p[j]);
  red[tid] = m;
  __syncthreads();
  for (int s = 128; s > 0; s >>= 1) {
    if (tid < s) red[tid] = fmaxf(red[tid], red[tid + s]);
    __syncthreads();
  }
  m = red[0];
  __syncthreads();

  float l = 0.0f;
  for (int j = tid; j < ncols; j += 256) l += __expf((float)p[j] - m);
  red[tid] = l;
  __syncthreads();
  for (int s = 128; s > 0; s >>= 1) {
    if (tid < s) red[tid] += red[tid + s];
    __syncthreads();
  }
  const float inv = 1.0f / red[0];

  for (int j = tid; j < ncols; j += 256)
    p[j] = (_Float16)(__expf((float)p[j] - m) * inv);
}

// ---------------------------------------------------------------------------
// Launch: wq = relu(Q Wq^T), wr = relu(R Wr^T), S = softmax(wq wr^T / sqrt(d)),
//         out = S @ R.
// Workspace layout (f16): wq[8192x2048] | wr[8192x2048] | refT[2048x8192] |
//                         S[8192x8192]   (~224 MB total)
// ---------------------------------------------------------------------------
extern "C" void kernel_launch(void* const* d_in, const int* in_sizes, int n_in,
                              void* d_out, int out_size, void* d_ws, size_t ws_size,
                              hipStream_t stream) {
  const float* query = (const float*)d_in[0];   // [8192, 2048]
  const float* ref   = (const float*)d_in[1];   // [8192, 2048]
  const float* Wq    = (const float*)d_in[2];   // [2048, 2048]
  const float* Wr    = (const float*)d_in[3];   // [2048, 2048]
  float* out = (float*)d_out;                   // [8192, 2048] fp32

  char* ws = (char*)d_ws;
  const size_t SZ_ACT = (size_t)NQ_ * DOUT_ * sizeof(_Float16);     // 32 MB
  _Float16* wq_h  = (_Float16*)(ws);
  _Float16* wr_h  = (_Float16*)(ws + SZ_ACT);
  _Float16* refT  = (_Float16*)(ws + 2 * SZ_ACT);                   // [2048, 8192]
  _Float16* S     = (_Float16*)(ws + 3 * SZ_ACT);                   // [8192, 8192]

  const dim3 b128(128);
  const float inv_sqrt_d = 0.022097086912079608f;   // 1/sqrt(2048)

  // wq = relu(query @ Wq^T)   M=8192 N=2048 K=2048
  gemm_nt_wmma<float, true, true>
      <<<dim3(NQ_ / 128, DOUT_ / 128), b128, 0, stream>>>(
          query, Wq, wq_h, NQ_, DOUT_, DQ_, 1.0f);

  // wr = relu(ref @ Wr^T)     M=8192 N=2048 K=2048
  gemm_nt_wmma<float, true, true>
      <<<dim3(NR_ / 128, DOUT_ / 128), b128, 0, stream>>>(
          ref, Wr, wr_h, NR_, DOUT_, DR_, 1.0f);

  // refT[c, j] = f16(ref[j, c])
  transpose_cvt<<<dim3(DR_ / 32, NR_ / 32), 256, 0, stream>>>(ref, refT, NR_, DR_);

  // S = (wq @ wr^T) / sqrt(d)  M=8192 N=8192 K=2048  (f16 logits)
  gemm_nt_wmma<_Float16, false, true>
      <<<dim3(NQ_ / 128, NR_ / 128), b128, 0, stream>>>(
          wq_h, wr_h, S, NQ_, NR_, DOUT_, inv_sqrt_d);

  // row-wise softmax in place
  softmax_rows<<<NQ_, 256, 0, stream>>>(S, NR_);

  // out = S @ refT^T           M=8192 N=2048 K=8192  (fp32 output)
  gemm_nt_wmma<_Float16, false, false>
      <<<dim3(NQ_ / 128, DR_ / 128), b128, 0, stream>>>(
          S, refT, out, NQ_, DR_, NR_, 1.0f);
}
